// GAT_54150947668287
// MI455X (gfx1250) — matrix operand order repaired and checked by reading, not compiled, not run on hardware
//
#include <hip/hip_runtime.h>
#include <math.h>

#define NN 8192
#define FF 1024
#define ALPHA 0.2f
#define NEGINF (-9.0e15f)

typedef __attribute__((ext_vector_type(16))) __bf16 v16bf;
typedef __attribute__((ext_vector_type(8)))  __bf16 v8bf;
typedef __attribute__((ext_vector_type(8)))  float  v8f;

union V16U { v16bf v; v8bf h[2]; };

// ---------------------------------------------------------------------------
// Kernel 1: cast X (fp32 row-major [N,F]) -> bf16 row-major, 8 elems/thread
// ---------------------------------------------------------------------------
__global__ __launch_bounds__(256) void k_cast_x(const float* __restrict__ X,
                                                __bf16* __restrict__ Xb) {
  int i = (blockIdx.x * 256 + threadIdx.x) * 8;
  float4 v0 = *(const float4*)(X + i);
  float4 v1 = *(const float4*)(X + i + 4);
  v8bf o;
  o[0] = (__bf16)v0.x; o[1] = (__bf16)v0.y; o[2] = (__bf16)v0.z; o[3] = (__bf16)v0.w;
  o[4] = (__bf16)v1.x; o[5] = (__bf16)v1.y; o[6] = (__bf16)v1.z; o[7] = (__bf16)v1.w;
  *(v8bf*)(Xb + i) = o;  // 16B aligned
}

// ---------------------------------------------------------------------------
// Kernel 2: W [F_IN, F_OUT] fp32 -> Wt bf16 [F_OUT, F_IN] (transposed so that
// WMMA B-fragments (lane = N, K contiguous) are single 32B vector loads)
// ---------------------------------------------------------------------------
__global__ __launch_bounds__(256) void k_cast_wt(const float* __restrict__ W,
                                                 __bf16* __restrict__ Wt) {
  int e = blockIdx.x * 256 + threadIdx.x;   // 1M elements
  int n = e >> 10;
  int k = e & 1023;
  Wt[n * FF + k] = (__bf16)W[k * FF + n];   // coalesced write
}

// ---------------------------------------------------------------------------
// Kernel 3: zero f1/f2 (gemm1 accumulates into them with atomics)
// ---------------------------------------------------------------------------
__global__ __launch_bounds__(256) void k_zero2(float* __restrict__ a,
                                               float* __restrict__ b) {
  int i = blockIdx.x * 256 + threadIdx.x;
  a[i] = 0.0f;
  b[i] = 0.0f;
}

// ---------------------------------------------------------------------------
// Kernel 4: h = X @ W via v_wmma_f32_16x16x32_bf16.
// Wave tile 32x64 (two row tiles share every B fragment: 2 WMMAs per
// B-load pair), K loop fully unrolled. Epilogue writes h^T bf16 and reduces
// f1 = h@a1, f2 = h@a2 via shuffles + global_atomic_add_f32.
// ---------------------------------------------------------------------------
__global__ __launch_bounds__(256) void k_gemm1(
    const __bf16* __restrict__ Xb, const __bf16* __restrict__ Wt,
    const float* __restrict__ a1, const float* __restrict__ a2,
    __bf16* __restrict__ ht, float* __restrict__ f1, float* __restrict__ f2) {
  const int lane = threadIdx.x & 31;
  const int wid  = threadIdx.x >> 5;
  const int w    = blockIdx.x * 8 + wid;       // 4096 waves total
  const int row0 = (w >> 4) * 32;              // 256 row tiles (32 rows each)
  const int col0 = (w & 15) * 64;              // 16 col groups
  const int half = lane >> 4;
  const int l15  = lane & 15;

  const __bf16* __restrict__ pA0row = Xb + (size_t)(row0 + l15) * FF + (half ? 16 : 0);
  const __bf16* __restrict__ pA1row = pA0row + (size_t)16 * FF;
  const __bf16* __restrict__ pBrow  = Wt + (size_t)(col0 + l15) * FF + (half ? 16 : 0);

  v8f c0[4] = {};   // rows row0..row0+15
  v8f c1[4] = {};   // rows row0+16..row0+31
#pragma unroll
  for (int kk = 0; kk < FF; kk += 32) {
    V16U A0, A1;
    A0.h[0] = *(const v8bf*)(pA0row + kk);        // 16B aligned
    A0.h[1] = *(const v8bf*)(pA0row + kk + 8);
    A1.h[0] = *(const v8bf*)(pA1row + kk);
    A1.h[1] = *(const v8bf*)(pA1row + kk + 8);
#pragma unroll
    for (int n = 0; n < 4; ++n) {
      const v16bf Bv = *(const v16bf*)(pBrow + (size_t)n * 16 * FF + kk);  // 32B aligned
      c0[n] = __builtin_amdgcn_wmma_f32_16x16x32_bf16(
          false, A0.v, false, Bv, (short)0, c0[n], false, false);
      c1[n] = __builtin_amdgcn_wmma_f32_16x16x32_bf16(
          false, A1.v, false, Bv, (short)0, c1[n], false, false);
    }
  }

  // a-vector values for this lane's 4 column positions
  float a1v[4], a2v[4];
#pragma unroll
  for (int n = 0; n < 4; ++n) {
    int col = col0 + n * 16 + l15;
    a1v[n] = a1[col];
    a2v[n] = a2[col];
  }

#pragma unroll
  for (int t = 0; t < 2; ++t) {
    v8f* c = t ? c1 : c0;
#pragma unroll
    for (int r = 0; r < 8; ++r) {
      const int row = row0 + t * 16 + r + half * 8;  // C/D: VGPR r -> M=r (+8 hi half)
      float t1 = 0.0f, t2 = 0.0f;
#pragma unroll
      for (int n = 0; n < 4; ++n) {
        float v = c[n][r];
        int col = col0 + n * 16 + l15;
        ht[(size_t)col * NN + row] = (__bf16)v;   // transposed bf16 store
        t1 += v * a1v[n];
        t2 += v * a2v[n];
      }
#pragma unroll
      for (int off = 8; off >= 1; off >>= 1) {    // reduce within 16-lane halves
        t1 += __shfl_xor(t1, off, 32);
        t2 += __shfl_xor(t2, off, 32);
      }
      if (l15 == 0) {
        atomicAdd(&f1[row], t1);
        atomicAdd(&f2[row], t2);
      }
    }
  }
}

// ---------------------------------------------------------------------------
// Kernel 5: per-row online softmax stats (max, 1/sum), reference mask
// semantics (-9e15 finite). One wave per row, adj streamed once. Two
// independent online accumulator pairs per lane break the serial exp chain.
// ---------------------------------------------------------------------------
__global__ __launch_bounds__(256) void k_stats(
    const int* __restrict__ adj, const float* __restrict__ f1,
    const float* __restrict__ f2, float* __restrict__ rowmax,
    float* __restrict__ rowinv) {
  const int lane = threadIdx.x & 31;
  const int wid  = threadIdx.x >> 5;
  const int row  = blockIdx.x * 8 + wid;
  const float fi = f1[row];
  const int* __restrict__ arow = adj + (size_t)row * NN;

  float m0 = -3.0e38f, s0 = 0.0f;
  float m1 = -3.0e38f, s1 = 0.0f;
  for (int j0 = lane * 4; j0 < NN; j0 += 128) {
    __builtin_prefetch(arow + j0 + 512, 0, 0);   // global_prefetch_b8
    int4 av   = *(const int4*)(arow + j0);
    float4 fv = *(const float4*)(f2 + j0);
    float xs[4] = {fv.x, fv.y, fv.z, fv.w};
    int   as[4] = {av.x, av.y, av.z, av.w};
#pragma unroll
    for (int t = 0; t < 4; ++t) {
      float x = fi + xs[t];
      x = x > 0.0f ? x : ALPHA * x;
      if (as[t] <= 0) x = NEGINF;
      if (t & 1) {
        float nm = fmaxf(m1, x);
        s1 = s1 * __expf(m1 - nm) + __expf(x - nm);
        m1 = nm;
      } else {
        float nm = fmaxf(m0, x);
        s0 = s0 * __expf(m0 - nm) + __expf(x - nm);
        m0 = nm;
      }
    }
  }
  // merge the two per-lane accumulators
  {
    float nm = fmaxf(m0, m1);
    s0 = s0 * __expf(m0 - nm) + s1 * __expf(m1 - nm);
    m0 = nm;
  }
  // cross-lane online combine
#pragma unroll
  for (int off = 16; off >= 1; off >>= 1) {
    float om = __shfl_xor(m0, off, 32);
    float os = __shfl_xor(s0, off, 32);
    float nm = fmaxf(m0, om);
    s0 = s0 * __expf(m0 - nm) + os * __expf(om - nm);
    m0 = nm;
  }
  if (lane == 0) {
    rowmax[row] = m0;
    rowinv[row] = 1.0f / s0;
  }
}

// ---------------------------------------------------------------------------
// Kernel 6: fused attention GEMM: out = relu(P @ h), P generated on the fly.
// Workgroup = 16 rows x 1024 features (adj streamed exactly once from HBM).
// 64-K strip per barrier pair: thread computes 4 probability entries into a
// 16x64 LDS strip (row stride 72 bf16: every ds_load_b128 16B-aligned, rows
// land on distinct bank groups), then two 8-WMMA blocks run per iteration.
// Only Ba is loop-carried; Bb and both A fragments are intra-step, and the
// next strip's adj/f2 inputs are prefetched before the WMMA blocks.
// ---------------------------------------------------------------------------
__global__ __launch_bounds__(256) void k_gemm2(
    const int* __restrict__ adj, const float* __restrict__ f1g,
    const float* __restrict__ f2g, const float* __restrict__ rowmax,
    const float* __restrict__ rowinv, const __bf16* __restrict__ ht,
    float* __restrict__ out) {
  __shared__ __bf16 pt[16 * 72];   // 16x64 strip, row stride 72 bf16 (144B)

  const int tid  = threadIdx.x;
  const int lane = tid & 31;
  const int wid  = tid >> 5;
  const int half = lane >> 4;
  const int l15  = lane & 15;
  const int row0 = blockIdx.x * 16;
  const int fbase = wid * 128;

  // P-generation mapping: thread produces rows r0,r0+8 at columns jj, jj+32
  const int r0 = tid >> 5;          // 0..7
  const int jj = tid & 31;          // 0..31
  const int r1 = r0 + 8;
  const float f1a = f1g[row0 + r0], f1b = f1g[row0 + r1];
  const float ma  = rowmax[row0 + r0], mb  = rowmax[row0 + r1];
  const float iza = rowinv[row0 + r0], izb = rowinv[row0 + r1];
  const int* __restrict__ arow0 = adj + (size_t)(row0 + r0) * NN;
  const int* __restrict__ arow1 = adj + (size_t)(row0 + r1) * NN;
  const __bf16* __restrict__ pBrow = ht + (size_t)(fbase + l15) * NN + (half ? 16 : 0);

  v8f acc[8] = {};
  v16bf Ba[8], Bb[8];

  // prologue: P inputs for strip k=0 and B fragments for tile k=0
  int   pa00 = arow0[jj], pa01 = arow0[jj + 32];
  int   pa10 = arow1[jj], pa11 = arow1[jj + 32];
  float pf0  = f2g[jj],   pf1  = f2g[jj + 32];
#pragma unroll
  for (int n = 0; n < 8; ++n)
    Ba[n] = *(const v16bf*)(pBrow + (size_t)n * 16 * NN);

  for (int k0 = 0; k0 < NN; k0 += 64) {
    // ---- build 16x64 probability strip from prefetched inputs ----
    {
      float x;
      x = f1a + pf0; x = x > 0.0f ? x : ALPHA * x; if (pa00 <= 0) x = NEGINF;
      pt[r0 * 72 + jj]      = (__bf16)(__expf(x - ma) * iza);
      x = f1a + pf1; x = x > 0.0f ? x : ALPHA * x; if (pa01 <= 0) x = NEGINF;
      pt[r0 * 72 + jj + 32] = (__bf16)(__expf(x - ma) * iza);
      x = f1b + pf0; x = x > 0.0f ? x : ALPHA * x; if (pa10 <= 0) x = NEGINF;
      pt[r1 * 72 + jj]      = (__bf16)(__expf(x - mb) * izb);
      x = f1b + pf1; x = x > 0.0f ? x : ALPHA * x; if (pa11 <= 0) x = NEGINF;
      pt[r1 * 72 + jj + 32] = (__bf16)(__expf(x - mb) * izb);
    }
    __syncthreads();

    // ---- A fragments for both 32-K tiles of the strip ----
    V16U A0, A1;
    const __bf16* pA = &pt[l15 * 72 + (half ? 16 : 0)];
    A0.h[0] = *(const v8bf*)pA;          // ds_load_b128, 16B aligned
    A0.h[1] = *(const v8bf*)(pA + 8);
    A1.h[0] = *(const v8bf*)(pA + 32);
    A1.h[1] = *(const v8bf*)(pA + 40);
    __syncthreads();

    // ---- prefetch next strip's P inputs ----
    const int kn = (k0 + 64) & (NN - 1);
    pa00 = arow0[kn + jj]; pa01 = arow0[kn + jj + 32];
    pa10 = arow1[kn + jj]; pa11 = arow1[kn + jj + 32];
    pf0  = f2g[kn + jj];   pf1  = f2g[kn + jj + 32];

    // ---- B(k0+32) in flight while WMMA block 1 runs on Ba = B(k0) ----
#pragma unroll
    for (int n = 0; n < 8; ++n)
      Bb[n] = *(const v16bf*)(pBrow + (size_t)n * 16 * NN + k0 + 32);
#pragma unroll
    for (int n = 0; n < 8; ++n)
      acc[n] = __builtin_amdgcn_wmma_f32_16x16x32_bf16(
          false, A0.v, false, Ba[n], (short)0, acc[n], false, false);

    // ---- B(k0+64) in flight while WMMA block 2 runs on Bb = B(k0+32) ----
#pragma unroll
    for (int n = 0; n < 8; ++n)
      Ba[n] = *(const v16bf*)(pBrow + (size_t)n * 16 * NN + kn);
#pragma unroll
    for (int n = 0; n < 8; ++n)
      acc[n] = __builtin_amdgcn_wmma_f32_16x16x32_bf16(
          false, A1.v, false, Bb[n], (short)0, acc[n], false, false);
  }

  // relu epilogue, C/D layout
#pragma unroll
  for (int n = 0; n < 8; ++n) {
#pragma unroll
    for (int r = 0; r < 8; ++r) {
      int row = row0 + r + half * 8;
      int col = fbase + n * 16 + l15;
      float v = acc[n][r];
      out[(size_t)row * FF + col] = v > 0.0f ? v : 0.0f;
    }
  }
}

// ---------------------------------------------------------------------------
extern "C" void kernel_launch(void* const* d_in, const int* in_sizes, int n_in,
                              void* d_out, int out_size, void* d_ws, size_t ws_size,
                              hipStream_t stream) {
  const float* X   = (const float*)d_in[0];   // [8192, 1024]
  const int*   adj = (const int*)d_in[1];     // [8192, 8192]
  const float* W   = (const float*)d_in[2];   // [1024, 1024]
  const float* a1  = (const float*)d_in[3];   // [1024]
  const float* a2  = (const float*)d_in[4];   // [1024]
  float* out = (float*)d_out;                 // [8192, 1024] fp32

  char* ws = (char*)d_ws;
  __bf16* Xb = (__bf16*)ws;                                  // 16 MB
  __bf16* Wt = (__bf16*)(ws + (size_t)16 * 1024 * 1024);     //  2 MB
  __bf16* ht = (__bf16*)(ws + (size_t)18 * 1024 * 1024);     // 16 MB
  float*  f1 = (float*)(ws + (size_t)34 * 1024 * 1024);      // 32 KB
  float*  f2   = f1 + NN;
  float*  rmax = f2 + NN;
  float*  rinv = rmax + NN;

  k_cast_x <<<(NN * FF / 8) / 256, 256, 0, stream>>>(X, Xb);
  k_cast_wt<<<(FF * FF) / 256,     256, 0, stream>>>(W, Wt);
  k_zero2  <<<NN / 256,            256, 0, stream>>>(f1, f2);
  k_gemm1  <<<512,  256, 0, stream>>>(Xb, Wt, a1, a2, ht, f1, f2);
  k_stats  <<<1024, 256, 0, stream>>>(adj, f1, f2, rmax, rinv);
  k_gemm2  <<<512,  256, 0, stream>>>(adj, f1, f2, rmax, rinv, ht, out);
}
